// SparseSelfAttention_37555194037029
// MI455X (gfx1250) — matrix operand order repaired and checked by reading, not compiled, not run on hardware
//
#include <hip/hip_runtime.h>

// Block-sparse causal-local attention for MI455X (gfx1250, wave32).
// Transposed flash formulation: S^T = K*Q^T, O^T = V^T*P^T, so each query row
// lives in one lane -> scalar softmax state, permlanex16 half-exchanges instead
// of ds_bpermute butterflies, and a pure-VALU P relayout (no LDS round trip).
// Key-padding bias enters as a rank-1 WMMA that initializes the S accumulators.
// B=2 H=16 L=2048 D=64 BLK=32 NB=64. One workgroup (2 waves) per 32-row Q block.

typedef __attribute__((ext_vector_type(16))) _Float16 v16h;
typedef __attribute__((ext_vector_type(8)))  float    v8f;
typedef __attribute__((ext_vector_type(4)))  float    f32x4;

#define BB   2
#define HH   16
#define LLEN 2048
#define DD   64
#define BLKS 32
#define NB   64
#define LOG2E 1.4426950408889634f

__device__ __forceinline__ v8f wmma_f16(v16h a, v16h b, v8f c) {
  // D = A(16x32 f16) * B(32x16 f16) + C(16x16 f32)
  return __builtin_amdgcn_wmma_f32_16x16x32_f16(false, a, false, b, (short)0, c,
                                                false, false);
}

// Exchange values between the two 16-lane halves (identity selector).
__device__ __forceinline__ float xhalf_f(float x) {
  const int i = __builtin_bit_cast(int, x);
  const int r = __builtin_amdgcn_permlanex16(i, i, 0x76543210, 0xFEDCBA98,
                                             false, false);
  return __builtin_bit_cast(float, r);
}

__global__ __launch_bounds__(64, 1) void sparse_attn_gfx1250(
    const float* __restrict__ Qg, const float* __restrict__ Kg,
    const float* __restrict__ Vg, const float* __restrict__ Pm,
    const int* __restrict__ Lay, float* __restrict__ Og) {
  // LDS: K row-major [key][d] f16, V transposed [d][key] f16.
  __shared__ _Float16 Klds[BLKS][DD + 8];
  __shared__ _Float16 Vlds[DD][BLKS + 8];

  const int gid = blockIdx.x;
  const int iq  = gid % NB;
  const int h   = (gid / NB) % HH;
  const int b   = gid / (NB * HH);

  const int tid  = threadIdx.x;
  const int lane = tid & 31;
  const int wid  = tid >> 5;   // wave 0: q-rows 0-15, wave 1: q-rows 16-31
  const int col  = lane & 15;  // C-layout column = this lane's query row
  const int kh   = lane >> 4;  // lane-half (C-layout row group)

  const size_t bh = (size_t)(b * HH + h) * LLEN;

  // ---- active-block bitmask from layout[h, iq, :] (uniform -> scalar) ----
  const int* lay = Lay + (h * NB + iq) * NB;
  unsigned long long mask = 0ull;
  for (int j = 0; j < NB; ++j)
    if (lay[j] != 0) mask |= (1ull << j);

  float* orow = Og + (bh + (size_t)(iq * BLKS + wid * 16 + col)) * DD;
  if (mask == 0ull) {  // no active keys: emit zeros
#pragma unroll
    for (int t = 0; t < 4; ++t) {
      f32x4 z = {};
      *(f32x4*)(orow + t * 16 + kh * 8)     = z;
      *(f32x4*)(orow + t * 16 + kh * 8 + 4) = z;
    }
    return;
  }

  // ---- Q^T -> B fragments (lane = q-row, d in AMD table order), pre-scaled ----
  const float qscale = 0.125f * LOG2E;  // D^-0.5 * log2(e)
  const float* qp = Qg + (bh + (size_t)(iq * BLKS + wid * 16 + col)) * DD;
  v16h bq[2];
#pragma unroll
  for (int ks = 0; ks < 2; ++ks)
#pragma unroll
    for (int e = 0; e < 16; ++e)
      bq[ks][e] = (_Float16)(qp[ks * 32 + kh * 8 + e + (e & 8)] * qscale);

  // ---- rank-1 helper: B tile of ones at k=0 (for kpm bias injection) ----
  v16h bone = {};
  if (kh == 0) bone[0] = (_Float16)1.0f;

  // ---- accumulators (O^T: 4 d-tiles) + scalar softmax state per lane ----
  v8f o[4] = {};
  float mrun = -3.0e38f, lrun = 0.0f;

  // ---- staging assignment: 64 threads cover 32 keys x 64 d ----
  const int skey = tid >> 1;
  const int sd   = (tid & 1) * 32;
  f32x4 kr[8], vr[8];
  auto load_regs = [&](int jb) {
    const float* kp = Kg + (bh + (size_t)(jb * BLKS + skey)) * DD + sd;
    const float* vp = Vg + (bh + (size_t)(jb * BLKS + skey)) * DD + sd;
#pragma unroll
    for (int i = 0; i < 8; ++i) {
      kr[i] = *(const f32x4*)(kp + i * 4);
      vr[i] = *(const f32x4*)(vp + i * 4);
    }
  };

  unsigned long long m = mask;
  int j = __ffsll(m) - 1;
  load_regs(j);

  while (true) {
    m &= (m - 1);
    const int jn = m ? (__ffsll(m) - 1) : -1;

    __syncthreads();  // previous compute done reading LDS tile
#pragma unroll
    for (int i = 0; i < 8; ++i)
#pragma unroll
      for (int c = 0; c < 4; ++c) {
        const int d = sd + i * 4 + c;
        Klds[skey][d] = (_Float16)kr[i][c];
        Vlds[d][skey] = (_Float16)vr[i][c];  // store V transposed
      }
    __syncthreads();  // tile ready

    if (jn >= 0) {
      load_regs(jn);  // hide next block's global latency behind compute
      const unsigned long long m2 = m & (m - 1);
      if (m2) {  // L2 prefetch two blocks ahead -> global_prefetch_b8
        const int j2 = __ffsll(m2) - 1;
        __builtin_prefetch(Kg + (bh + (size_t)(j2 * BLKS + skey)) * DD + sd, 0, 1);
        __builtin_prefetch(Vg + (bh + (size_t)(j2 * BLKS + skey)) * DD + sd, 0, 1);
      }
    }

    // ---- S^T init = kpm bias (rank-1 WMMA: kpm column x ones row) ----
    const float kpm0 = Pm[b * LLEN + j * BLKS + col] * LOG2E;
    const float kpm1 = Pm[b * LLEN + j * BLKS + 16 + col] * LOG2E;
    v16h akp0 = {}, akp1 = {};
    if (kh == 0) { akp0[0] = (_Float16)kpm0; akp1[0] = (_Float16)kpm1; }
    const v8f zacc = {};
    v8f s0 = wmma_f16(akp0, bone, zacc);  // key tile 0 (keys 0-15)
    v8f s1 = wmma_f16(akp1, bone, zacc);  // key tile 1 (keys 16-31)

    // ---- S^T += K @ Q^T (A = K tiles from LDS, 2 k-steps over D=64) ----
#pragma unroll
    for (int ks = 0; ks < 2; ++ks) {
      v16h ka0, ka1;
      const _Float16* p0 = &Klds[col][ks * 32 + kh * 8];       // key = col
      const _Float16* p1 = &Klds[16 + col][ks * 32 + kh * 8];  // key = 16+col
#pragma unroll
      for (int e = 0; e < 8; ++e) {
        ka0[e] = p0[e]; ka0[8 + e] = p0[16 + e];  // A swizzle: two 8-runs
        ka1[e] = p1[e]; ka1[8 + e] = p1[16 + e];
      }
      s0 = wmma_f16(ka0, bq[ks], s0);
      s1 = wmma_f16(ka1, bq[ks], s1);
    }

    // ---- online softmax: this lane owns one query row ----
    float mloc = s0[0];
#pragma unroll
    for (int r = 1; r < 8; ++r) mloc = fmaxf(mloc, s0[r]);
#pragma unroll
    for (int r = 0; r < 8; ++r) mloc = fmaxf(mloc, s1[r]);
    mloc = fmaxf(mloc, xhalf_f(mloc));  // other half holds the other 16 keys
    const float mnew  = fmaxf(mrun, mloc);
    const float alpha = __builtin_amdgcn_exp2f(mrun - mnew);

    float ls = 0.0f;
#pragma unroll
    for (int r = 0; r < 8; ++r) {
      s0[r] = __builtin_amdgcn_exp2f(s0[r] - mnew);
      s1[r] = __builtin_amdgcn_exp2f(s1[r] - mnew);
      ls += s0[r] + s1[r];
    }
    ls += xhalf_f(ls);
    lrun = lrun * alpha + ls;
    mrun = mnew;
#pragma unroll
    for (int t = 0; t < 4; ++t)
#pragma unroll
      for (int r = 0; r < 8; ++r) o[t][r] *= alpha;

    // ---- P^T (C layout) -> B fragment: pure-VALU half-swap relayout ----
    // target elem e   (keys kh*16+e, e<8):  lower<-s0[e],       upper<-x(s1[e])
    // target elem 8+e (keys kh*16+8+e):     lower<-x(s0[e]),    upper<-s1[e]
    v16h pb;
#pragma unroll
    for (int e = 0; e < 8; ++e) {
      const float lo = s0[e], hi = s1[e];
      pb[e]     = (_Float16)((kh == 0) ? lo : xhalf_f(hi));
      pb[8 + e] = (_Float16)((kh == 0) ? xhalf_f(lo) : hi);
    }

    // ---- O^T += V^T @ P^T (A = V^T tiles from LDS) ----
#pragma unroll
    for (int t = 0; t < 4; ++t) {
      v16h va;
      const _Float16* vp2 = &Vlds[t * 16 + col][kh * 16];
#pragma unroll
      for (int e = 0; e < 8; ++e) { va[e] = vp2[e]; va[8 + e] = vp2[8 + e]; }
      o[t] = wmma_f16(va, pb, o[t]);
    }

    if (jn < 0) break;
    j = jn;
  }

  // ---- normalize and store: each lane owns one row, contiguous d runs ----
  const float rl = 1.0f / lrun;
#pragma unroll
  for (int t = 0; t < 4; ++t) {
    f32x4 w0 = {o[t][0] * rl, o[t][1] * rl, o[t][2] * rl, o[t][3] * rl};
    f32x4 w1 = {o[t][4] * rl, o[t][5] * rl, o[t][6] * rl, o[t][7] * rl};
    *(f32x4*)(orow + t * 16 + kh * 8)     = w0;
    *(f32x4*)(orow + t * 16 + kh * 8 + 4) = w1;
  }
}

extern "C" void kernel_launch(void* const* d_in, const int* in_sizes, int n_in,
                              void* d_out, int out_size, void* d_ws,
                              size_t ws_size, hipStream_t stream) {
  const float* Q   = (const float*)d_in[0];
  const float* K   = (const float*)d_in[1];
  const float* V   = (const float*)d_in[2];
  const float* Pmk = (const float*)d_in[3];
  const int*   Lay = (const int*)d_in[4];
  float* O = (float*)d_out;
  (void)in_sizes; (void)n_in; (void)out_size; (void)d_ws; (void)ws_size;

  dim3 grid(BB * HH * NB);  // 2048 workgroups: one per (b, h, q-block)
  dim3 block(64);           // 2 wave32s
  sparse_attn_gfx1250<<<grid, block, 0, stream>>>(Q, K, V, Pmk, Lay, O);
}
